// GNN_Encoder_73212012528430
// MI455X (gfx1250) — matrix operand order repaired and checked by reading, not compiled, not run on hardware
//
#include <hip/hip_runtime.h>

// ---------------------------------------------------------------------------
// GCN encoder for MI455X (gfx1250, wave32, WMMA).
// Memory-bound (~132 MB @ 23.3 TB/s => ~5.7us floor). All intermediates in
// LDS; matmuls via v_wmma_f32_16x16x32_f16. Aggregation and FC GEMMs are
// computed transposed (h^T = z^T * Ahat^T) so every writeback is a packed
// ds_store_b128; biases ride in the WMMA C operand; fast tanh via v_exp/v_rcp.
// ---------------------------------------------------------------------------

typedef _Float16 v16h __attribute__((ext_vector_type(16)));
typedef _Float16 v8h  __attribute__((ext_vector_type(8)));
typedef _Float16 v4h  __attribute__((ext_vector_type(4)));
typedef float    v8f  __attribute__((ext_vector_type(8)));

#define NUM_AREA  63
#define NUM_EDGES 252
#define XROW      756   // 63*12
#define OROW      252   // 63*4

#define LDS_FENCE() asm volatile("s_wait_dscnt 0" ::: "memory")

#define WMMA(a, b, c) \
  __builtin_amdgcn_wmma_f32_16x16x32_f16(false, (a), false, (b), (short)0, (c), false, false)

__device__ __forceinline__ float fast_tanh(float x) {
  // tanh(x) = sign(x) * (1 - e) / (1 + e),  e = exp(-2|x|)
  float e = __builtin_amdgcn_exp2f(-2.885390082f * fabsf(x));
  float r = (1.f - e) * __builtin_amdgcn_rcpf(1.f + e);
  return copysignf(r, x);
}

// A-operand fragment: 16x32 f16 tile from a row-major LDS matrix.
// Per ISA 7.12.2: lane<16 holds K = 0..7,16..23 ; lane>=16 holds K = 8..15,24..31
__device__ __forceinline__ v16h load_afrag(const _Float16* base, int row0, int kbase,
                                           int stride, int lane) {
  int m = lane & 15, hi = (lane >> 4) & 1;
  const _Float16* p = base + (row0 + m) * stride + kbase + hi * 8;
  v8h lo = *(const v8h*)p;
  v8h hg = *(const v8h*)(p + 16);
  return __builtin_shufflevector(lo, hg, 0,1,2,3,4,5,6,7,8,9,10,11,12,13,14,15);
}

// B-operand fragment: 32x16 f16 tile from a column-major (NxK) LDS matrix.
// lane holds column n = lane&15; lanes<16: K = 0..7,16..23 ; lanes>=16: K = 8..15,24..31
__device__ __forceinline__ v16h load_bfrag(const _Float16* base, int cstride, int kbase,
                                           int lane) {
  int n = lane & 15, hi = (lane >> 4) & 1;
  const _Float16* p = base + n * cstride + kbase + hi * 8;
  v8h lo = *(const v8h*)p;
  v8h hg = *(const v8h*)(p + 16);
  return __builtin_shufflevector(lo, hg, 0,1,2,3,4,5,6,7,8,9,10,11,12,13,14,15);
}

__global__ __launch_bounds__(256)
void gnn_encoder_kernel(const float* __restrict__ x, const long long* __restrict__ ei,
                        const float* __restrict__ rej, const float* __restrict__ theta,
                        const float* __restrict__ W1, const float* __restrict__ b1v,
                        const float* __restrict__ W2, const float* __restrict__ b2v,
                        const float* __restrict__ fcw, const float* __restrict__ fcb,
                        float* __restrict__ out, int batch) {
  // LDS layout (bytes):
  //   [0,8192)      Ahat   f16 64x64 row-major (== Ahat^T column-major, B operand)
  //   [8192,9216)   W1pad  f16 16x32 col-major (rows 16..27 = W1)
  //   [9216,10240)  W2pad  f16 16x32 col-major (rows 0..7  = W2)
  //   [10240,11264) FCpad  f16 16x32 (== fc_w^T row-major, A operand; rows 0..3 live)
  //   [11264,11520) dinv   f32 64
  //   [11520,60672) 8 per-wave buffers of 6144B:
  //                   bufA 64x32 f16 row-major (4096B): cols0..15 h, 16..27 x, 28..29 gf
  //                   bufB 16x64 f16 (z col-major == z^T row-major) (2048B)
  //   Af32 (f32 64x64 build scratch) aliases the wave-buffer region.
  __shared__ __align__(32) unsigned char smem[60672];
  _Float16* Ahat = (_Float16*)(smem);
  _Float16* W1p  = (_Float16*)(smem + 8192);
  _Float16* W2p  = (_Float16*)(smem + 9216);
  _Float16* FCp  = (_Float16*)(smem + 10240);
  float*    dinv = (float*)(smem + 11264);
  float*    Af32 = (float*)(smem + 11520);

  const int tid  = threadIdx.x;
  const int lane = tid & 31;
  const int wid  = tid >> 5;
  unsigned char* wb = smem + 11520 + wid * 6144;
  _Float16* bufA = (_Float16*)wb;
  _Float16* bufB = (_Float16*)(wb + 4096);

  // ---- one-time per-workgroup: A-hat + packed weights ----
  for (int i = tid; i < 4096; i += 256) Af32[i] = 0.f;
  for (int i = tid; i < 512; i += 256) {
    W1p[i] = (_Float16)0.f; W2p[i] = (_Float16)0.f; FCp[i] = (_Float16)0.f;
  }
  if (tid < 64) dinv[tid] = 0.f;
  __syncthreads();

  if (tid < NUM_AREA) {
    int d = 1;  // self loop
    for (int e = 0; e < NUM_EDGES; ++e) d += ((int)ei[NUM_EDGES + e] == tid);
    dinv[tid] = rsqrtf((float)d);
  }
  if (tid < 96)  { int k = tid >> 3, n = tid & 7;  W1p[n * 32 + 16 + k] = (_Float16)W1[k * 8  + n]; }
  if (tid < 128) { int k = tid >> 4, n = tid & 15; W2p[n * 32 + k]      = (_Float16)W2[k * 16 + n]; }
  if (tid < 120) { int k = tid >> 2, n = tid & 3;  FCp[n * 32 + k]      = (_Float16)fcw[k * 4 + n]; }
  __syncthreads();

  if (tid < NUM_EDGES) {
    int s = (int)ei[tid], t = (int)ei[NUM_EDGES + tid];
    atomicAdd(&Af32[t * 64 + s], dinv[s] * dinv[t]);     // ds_add_f32
  }
  if (tid < NUM_AREA) atomicAdd(&Af32[tid * 64 + tid], dinv[tid] * dinv[tid]);
  __syncthreads();
  for (int i = tid; i < 4096; i += 256) Ahat[i] = (_Float16)Af32[i];
  __syncthreads();

  // zero this wave's buffers (overwrites Af32 scratch region)
  { unsigned int* z = (unsigned int*)wb;
    for (int i = lane; i < 1536; i += 32) z[i] = 0u; }
  LDS_FENCE();

  const int n  = lane & 15;
  const int hi = (lane >> 4) & 1;

  // biases ride in the WMMA C operand (accumulator rows = output features)
  v8f bv1, bv2, bvf;
#pragma unroll
  for (int i = 0; i < 8; ++i) {
    bv1[i] = hi ? 0.f : b1v[i];          // h1 features 0..7, pad rows 0
    bv2[i] = b2v[i + hi * 8];            // h2 features 0..15
    bvf[i] = (hi == 0 && i < 4) ? fcb[i] : 0.f;
  }

  const v16h w1f = load_bfrag(W1p, 32, 0, lane);
  const v16h w2f = load_bfrag(W2p, 32, 0, lane);
  const v16h fcA = load_afrag(FCp, 0, 0, 32, lane);   // fc_w^T as A operand

  const int gw = blockIdx.x * 8 + wid;
  const int nw = gridDim.x * 8;

  for (int b = gw; b < batch; b += nw) {
    // ---- stage x (cols 16..27) and global features (cols 28..29) into bufA ----
    // 12 = 3*4, so every float4 of a row stays inside that row.
    const float4* xb = (const float4*)(x + (size_t)b * XROW);
#pragma unroll
    for (int j = 0; j < 6; ++j) {
      int q = lane + j * 32;
      if (q < 189) {
        float4 v = xb[q];
        int r = q / 3;
        int c = (q - r * 3) * 4;   // 0,4,8
        v4h h4;
        h4[0] = (_Float16)v.x; h4[1] = (_Float16)v.y;
        h4[2] = (_Float16)v.z; h4[3] = (_Float16)v.w;
        *(v4h*)(bufA + r * 32 + 16 + c) = h4;
      }
    }
    {
      union { _Float16 h[2]; unsigned int u; } cv;
      cv.h[0] = (_Float16)rej[b]; cv.h[1] = (_Float16)theta[b];
      for (int r = lane; r < NUM_AREA; r += 32)
        *(unsigned int*)(bufA + r * 32 + 28) = cv.u;
    }
    LDS_FENCE();

    // ---- z1 = X @ W1  -> bufB (z^T row-major) ----
    for (int mt = 0; mt < 4; ++mt) {
      v16h a = load_afrag(bufA, mt * 16, 0, 32, lane);
      v8f c = {};
      c = WMMA(a, w1f, c);
      v8h zv;
#pragma unroll
      for (int i = 0; i < 8; ++i) zv[i] = (_Float16)c[i];
      *(v8h*)(bufB + n * 64 + mt * 16 + hi * 8) = zv;
    }
    LDS_FENCE();

    // ---- h1^T = z1^T @ Ahat^T + b1, leaky -> bufA cols 0..15 (b128 stores) ----
    {
      v16h az0 = load_afrag(bufB, 0, 0,  64, lane);   // nt-invariant
      v16h az1 = load_afrag(bufB, 0, 32, 64, lane);
      for (int nt = 0; nt < 4; ++nt) {
        v16h bb0 = load_bfrag(Ahat + nt * 1024, 64, 0,  lane);
        v16h bb1 = load_bfrag(Ahat + nt * 1024, 64, 32, lane);
        v8f c = bv1;
        c = WMMA(az0, bb0, c);
        c = WMMA(az1, bb1, c);
        v8h hv;
#pragma unroll
        for (int i = 0; i < 8; ++i) {
          float v = fmaxf(c[i], 0.1f * c[i]);   // leaky_relu(0.1)
          hv[i] = (_Float16)v;
        }
        *(v8h*)(bufA + (nt * 16 + n) * 32 + hi * 8) = hv;  // node-row, 8 contiguous feats
      }
    }
    LDS_FENCE();

    // ---- z2 = h1 @ W2 -> bufB ----
    for (int mt = 0; mt < 4; ++mt) {
      v16h a = load_afrag(bufA, mt * 16, 0, 32, lane);
      v8f c = {};
      c = WMMA(a, w2f, c);
      v8h zv;
#pragma unroll
      for (int i = 0; i < 8; ++i) zv[i] = (_Float16)c[i];
      *(v8h*)(bufB + n * 64 + mt * 16 + hi * 8) = zv;
    }
    LDS_FENCE();

    // ---- h2^T = z2^T @ Ahat^T + b2, leaky -> bufA cols 0..15 ----
    {
      v16h az0 = load_afrag(bufB, 0, 0,  64, lane);
      v16h az1 = load_afrag(bufB, 0, 32, 64, lane);
      for (int nt = 0; nt < 4; ++nt) {
        v16h bb0 = load_bfrag(Ahat + nt * 1024, 64, 0,  lane);
        v16h bb1 = load_bfrag(Ahat + nt * 1024, 64, 32, lane);
        v8f c = bv2;
        c = WMMA(az0, bb0, c);
        c = WMMA(az1, bb1, c);
        v8h hv;
#pragma unroll
        for (int i = 0; i < 8; ++i) {
          float v = fmaxf(c[i], 0.1f * c[i]);
          hv[i] = (_Float16)v;
        }
        *(v8h*)(bufA + (nt * 16 + n) * 32 + hi * 8) = hv;
      }
    }
    LDS_FENCE();

    // ---- out^T = fc_w^T @ [h2|x|gf]^T + fc_b, tanh; stage f32 then coalesce ----
    float* ost = (float*)bufB;
    for (int nt = 0; nt < 4; ++nt) {
      v16h bb = load_bfrag(bufA + nt * 512, 32, 0, lane);  // concat rows as B operand
      v8f c = bvf;
      c = WMMA(fcA, bb, c);
      if (hi == 0) {               // rows 0..3 = the 4 outputs of node nt*16+n
        float4 o;
        o.x = fast_tanh(c[0]); o.y = fast_tanh(c[1]);
        o.z = fast_tanh(c[2]); o.w = fast_tanh(c[3]);
        *(float4*)(ost + (nt * 16 + n) * 4) = o;
      }
    }
    LDS_FENCE();
    {
      const float4* ost4 = (const float4*)ost;
      float4* ob = (float4*)(out + (size_t)b * OROW);   // b*1008 bytes, 16B aligned
      for (int i = lane; i < 63; i += 32) ob[i] = ost4[i];
    }
    LDS_FENCE();
  }
}

extern "C" void kernel_launch(void* const* d_in, const int* in_sizes, int n_in,
                              void* d_out, int out_size, void* d_ws, size_t ws_size,
                              hipStream_t stream) {
  (void)n_in; (void)out_size; (void)d_ws; (void)ws_size;
  const float*     x   = (const float*)d_in[0];
  const long long* ei  = (const long long*)d_in[1];   // int64 edge_index [2,252]
  const float*     rej = (const float*)d_in[3];
  const float*     th  = (const float*)d_in[4];
  const float*     W1  = (const float*)d_in[5];
  const float*     b1  = (const float*)d_in[6];
  const float*     W2  = (const float*)d_in[7];
  const float*     b2  = (const float*)d_in[8];
  const float*     fcw = (const float*)d_in[9];
  const float*     fcb = (const float*)d_in[10];
  float* out = (float*)d_out;
  int batch = in_sizes[0] / XROW;   // 32768

  dim3 grid(512), block(256);       // 4096 waves, 8 batch elements each
  gnn_encoder_kernel<<<grid, block, 0, stream>>>(x, ei, rej, th, W1, b1, W2, b2,
                                                 fcw, fcb, out, batch);
}
